// RoPEAttention_56719338111027
// MI455X (gfx1250) — matrix operand order repaired and checked by reading, not compiled, not run on hardware
//
#include <hip/hip_runtime.h>

typedef __bf16 bf16;
typedef __attribute__((ext_vector_type(8)))  __bf16 bf16x8;
typedef __attribute__((ext_vector_type(16))) __bf16 bf16x16;
typedef __attribute__((ext_vector_type(8)))  float  f32x8;

#define CDIM 1024
#define NCTX 2048
#define HHEADS 16
#define HD 64

union BF16x16 { bf16x16 v; bf16x8 h[2]; };

// A fragment (16x32 bf16, row-major source with leading dim ld).
// ISA 7.12.2: lane m=lane&15 holds row M=m; lane-half h: VGPR0-3 = K=8h..8h+7,
// VGPR4-7 = K=16+8h..16+8h+7  -> two contiguous 16B (b128) loads per lane.
__device__ __forceinline__ bf16x16 frag_a(const bf16* base, int row0, int ld, int kbase, int lane) {
  const int m = lane & 15, h = lane >> 4;
  const bf16* p = base + (size_t)(row0 + m) * ld + kbase + 8 * h;
  BF16x16 r;
  r.h[0] = *(const bf16x8*)(p);
  r.h[1] = *(const bf16x8*)(p + 16);
  return r.v;
}

// B fragment (32xK16 bf16) from an N-major LDS tile bt[n][k] (leading dim ld).
// Lane n=lane&15 holds column N=n; half h holds K=8h..8h+7 and K=16+8h..16+8h+7.
__device__ __forceinline__ bf16x16 frag_b(const bf16* bt, int col0, int ld, int kbase, int lane) {
  const int n = lane & 15, h = lane >> 4;
  const bf16* p = bt + (size_t)(col0 + n) * ld + kbase + 8 * h;
  BF16x16 r;
  r.h[0] = *(const bf16x8*)(p);
  r.h[1] = *(const bf16x8*)(p + 16);
  return r.v;
}

__device__ __forceinline__ f32x8 wmma_bf16(bf16x16 a, bf16x16 b, f32x8 c) {
  return __builtin_amdgcn_wmma_f32_16x16x32_bf16(false, a, false, b, (short)0, c, false, false);
}

// Async global->LDS 16B copy (gfx1250 GVS form), tracked by ASYNCcnt.
__device__ __forceinline__ void async_b128(const bf16* lds_dst, const bf16* gbase, uint32_t gbyteoff) {
  const uint32_t lds_addr = (uint32_t)(uintptr_t)lds_dst;   // low 32 bits = LDS offset
  asm volatile("global_load_async_to_lds_b128 %0, %1, %2"
               :: "v"(lds_addr), "v"(gbyteoff), "s"((uint64_t)(uintptr_t)gbase)
               : "memory");
}

// ---------------- Kernel 1: QKV GEMM (+bias) -> bf16 Q,K:[B][H][N][HD], V:[B][H][HD][N] ----------------
__global__ __launch_bounds__(256) void k_qkv_gemm(
    const float* __restrict__ x, const float* __restrict__ W, const float* __restrict__ bias,
    bf16* __restrict__ Qo, bf16* __restrict__ Ko, bf16* __restrict__ Vo)
{
  __shared__ __align__(16) bf16 As[128][32];   // row-major
  __shared__ __align__(16) bf16 Bs[128][32];   // N-major: Bs[col][k]
  const int tid = threadIdx.x, lane = tid & 31, wave = tid >> 5;
  const int wr = wave >> 2, wc = wave & 3;     // 2x4 wave grid
  const int m0 = blockIdx.x * 128;
  const int n0 = blockIdx.y * 128;

  f32x8 acc[4][2] = {};

  const int ar = tid >> 3, ac = (tid & 7) * 4;
  const int bkr = tid >> 5, bc = (tid & 31) * 4;

#pragma unroll 2
  for (int k0 = 0; k0 < CDIM; k0 += 32) {
    if (k0 + 32 < CDIM) {
      __builtin_prefetch(&x[(size_t)(m0 + ar) * CDIM + k0 + 32 + ac], 0, 1);
      __builtin_prefetch(&W[(size_t)(k0 + 32 + bkr) * (3 * CDIM) + n0 + bc], 0, 1);
    }
#pragma unroll
    for (int p = 0; p < 4; ++p) {
      const float4 f = *(const float4*)&x[(size_t)(m0 + ar + 32 * p) * CDIM + k0 + ac];
      bf16* d = &As[ar + 32 * p][ac];
      d[0] = (bf16)f.x; d[1] = (bf16)f.y; d[2] = (bf16)f.z; d[3] = (bf16)f.w;
    }
#pragma unroll
    for (int p = 0; p < 4; ++p) {
      const int kk = bkr + 8 * p;
      const float4 f = *(const float4*)&W[(size_t)(k0 + kk) * (3 * CDIM) + n0 + bc];
      Bs[bc + 0][kk] = (bf16)f.x; Bs[bc + 1][kk] = (bf16)f.y;
      Bs[bc + 2][kk] = (bf16)f.z; Bs[bc + 3][kk] = (bf16)f.w;
    }
    __syncthreads();
    const bf16x16 b0 = frag_b(&Bs[0][0], wc * 32 + 0, 32, 0, lane);
    const bf16x16 b1 = frag_b(&Bs[0][0], wc * 32 + 16, 32, 0, lane);
#pragma unroll
    for (int mt = 0; mt < 4; ++mt) {
      const bf16x16 a = frag_a(&As[0][0], wr * 64 + mt * 16, 32, 0, lane);
      acc[mt][0] = wmma_bf16(a, b0, acc[mt][0]);
      acc[mt][1] = wmma_bf16(a, b1, acc[mt][1]);
    }
    __syncthreads();
  }

  // Epilogue: bias + scatter. Q,K row-major per head; V transposed per head.
  const int h = lane >> 4;
#pragma unroll
  for (int mt = 0; mt < 4; ++mt)
#pragma unroll
    for (int nt = 0; nt < 2; ++nt)
#pragma unroll
      for (int i = 0; i < 8; ++i) {
        const int row = m0 + wr * 64 + mt * 16 + i + 8 * h;
        const int col = n0 + wc * 32 + nt * 16 + (lane & 15);
        const float v = acc[mt][nt][i] + bias[col];
        const int sel = col >> 10;
        const int ch = col & (CDIM - 1);
        const int hh = ch >> 6, hd = ch & 63;
        const int bb = row >> 11, nn = row & (NCTX - 1);
        size_t off;
        bf16* dst;
        if (sel == 2) {  // V: [B][H][HD][N]
          off = (((size_t)bb * HHEADS + hh) * HD + hd) * NCTX + nn;
          dst = Vo;
        } else {         // Q,K: [B][H][N][HD]
          off = (((size_t)bb * HHEADS + hh) * NCTX + nn) * HD + hd;
          dst = sel == 0 ? Qo : Ko;
        }
        dst[off] = (bf16)v;
      }
}

// ---------------- Kernel 2: RoPE in-place on Q and K ----------------
__global__ __launch_bounds__(256) void k_rope(bf16* __restrict__ Qo, bf16* __restrict__ Ko,
                                              const float* __restrict__ cosf,
                                              const float* __restrict__ sinf)
{
  const int idx = blockIdx.x * blockDim.x + threadIdx.x;   // pair index
  bf16* T = blockIdx.y == 0 ? Qo : Ko;
  const int i = idx & 31;                  // half-dim index (HD/2 = 32)
  const int n = (idx >> 5) & (NCTX - 1);
  const int bh = idx >> 16;                // 32*2048 = 65536 pairs per (b,h)
  const size_t off = ((size_t)bh * NCTX + n) * HD + 2 * i;
  const float c = cosf[n * 32 + i], s = sinf[n * 32 + i];
  const float t0 = (float)T[off], t1 = (float)T[off + 1];
  T[off]     = (bf16)(t0 * c - t1 * s);
  T[off + 1] = (bf16)(t0 * s + t1 * c);
}

// ---------------- Kernel 3: flash attention per (b,h), async double-buffered K/V ----------------
__global__ __launch_bounds__(256) void k_attn(const bf16* __restrict__ Q, const bf16* __restrict__ K,
                                              const bf16* __restrict__ Vt_g, bf16* __restrict__ O)
{
  __shared__ __align__(16) bf16 Qs[128][64];      // 16 KB
  __shared__ __align__(16) bf16 Ks[2][32][64];    // 8 KB  [key][hd], double-buffered
  __shared__ __align__(16) bf16 Vt[2][64][32];    // 8 KB  [hd][key], double-buffered
  __shared__ __align__(16) bf16 Ps[8][16][32];    // 8 KB  per-wave P scratch
  const int tid = threadIdx.x, lane = tid & 31, wave = tid >> 5;
  const int h = lane >> 4;
  const int bh = blockIdx.y;
  const size_t base = (size_t)bh * NCTX * HD;     // head base (elements) for Q/K and Vt
  const int q0 = blockIdx.x * 128;

  const bf16* Qh  = Q + base;
  const bf16* Kh  = K + base;
  const bf16* Vth = Vt_g + base;                  // [HD][NCTX] per head

  // Issue one 32-key tile (K rows + V^T rows) as async b128 copies: 2 instrs/thread.
  auto issue_tile = [&](int kt, int buf) {
    const int kr = tid >> 3, kc = (tid & 7) * 8;            // K: 32x64
    async_b128(&Ks[buf][kr][kc], Kh, (uint32_t)(((size_t)(kt + kr) * HD + kc) * 2));
    const int vr = tid >> 2, vc = (tid & 3) * 8;            // V^T: 64x32
    async_b128(&Vt[buf][vr][vc], Vth, (uint32_t)(((size_t)vr * NCTX + kt + vc) * 2));
  };

  // Prologue: async-stage Q block and first K/V tile.
  {
    const int r = tid >> 3, c = (tid & 7) * 8;
#pragma unroll
    for (int p = 0; p < 4; ++p)
      async_b128(&Qs[r + 32 * p][c], Qh, (uint32_t)(((size_t)(q0 + r + 32 * p) * HD + c) * 2));
  }
  issue_tile(0, 0);
  asm volatile("s_wait_asynccnt 0x0" ::: "memory");
  __syncthreads();

  const bf16x16 aq0 = frag_a(&Qs[0][0], wave * 16, 64, 0, lane);   // hd 0..31
  const bf16x16 aq1 = frag_a(&Qs[0][0], wave * 16, 64, 32, lane);  // hd 32..63

  f32x8 oacc[4] = {};
  float m_i[8], l_i[8];
#pragma unroll
  for (int i = 0; i < 8; ++i) { m_i[i] = -3.0e38f; l_i[i] = 0.f; }

#pragma unroll 2
  for (int kt = 0; kt < NCTX; kt += 32) {
    const int buf = (kt >> 5) & 1;
    // Overlap: stream tile kt+32 into the other buffer while computing on `buf`.
    if (kt + 32 < NCTX) {
      issue_tile(kt + 32, buf ^ 1);
      asm volatile("s_wait_asynccnt 0x2" ::: "memory");  // tile kt done; kt+32 in flight
    } else {
      asm volatile("s_wait_asynccnt 0x0" ::: "memory");
    }
    __syncthreads();

    // S = Q K^T for 32 keys: two 16x16 f32 tiles, K-dim = HD = 64 (2 steps)
    f32x8 s0 = {}, s1 = {};
    bf16x16 b;
    b = frag_b(&Ks[buf][0][0], 0, 64, 0, lane);   s0 = wmma_bf16(aq0, b, s0);
    b = frag_b(&Ks[buf][0][0], 0, 64, 32, lane);  s0 = wmma_bf16(aq1, b, s0);
    b = frag_b(&Ks[buf][0][0], 16, 64, 0, lane);  s1 = wmma_bf16(aq0, b, s1);
    b = frag_b(&Ks[buf][0][0], 16, 64, 32, lane); s1 = wmma_bf16(aq1, b, s1);

    // Online softmax. C/D layout: VGPR i holds (row=i+8h, col=lane&15); rows
    // live across 16-lane halves -> width-16 shuffle reductions.
#pragma unroll
    for (int i = 0; i < 8; ++i) {
      const float sc = 0.125f;                 // HD^-0.5
      const float v0 = s0[i] * sc, v1 = s1[i] * sc;
      float mx = fmaxf(v0, v1);
#pragma unroll
      for (int msk = 8; msk >= 1; msk >>= 1)
        mx = fmaxf(mx, __shfl_xor(mx, msk, 16));
      const float mnew = fmaxf(m_i[i], mx);
      const float alpha = __expf(m_i[i] - mnew);
      const float p0 = __expf(v0 - mnew);
      const float p1 = __expf(v1 - mnew);
      float rs = p0 + p1;
#pragma unroll
      for (int msk = 8; msk >= 1; msk >>= 1)
        rs += __shfl_xor(rs, msk, 16);
      l_i[i] = l_i[i] * alpha + rs;
      m_i[i] = mnew;
#pragma unroll
      for (int t = 0; t < 4; ++t) oacc[t][i] *= alpha;
      Ps[wave][i + 8 * h][(lane & 15)]      = (bf16)p0;
      Ps[wave][i + 8 * h][16 + (lane & 15)] = (bf16)p1;
    }
    // Wave-private LDS round-trip (C-layout -> A-layout); LDS in-order per wave.
    asm volatile("s_wait_dscnt 0" ::: "memory");

    const bf16x16 ap = frag_a(&Ps[wave][0][0], 0, 32, 0, lane);
#pragma unroll
    for (int dt = 0; dt < 4; ++dt) {
      const bf16x16 bv = frag_b(&Vt[buf][0][0], dt * 16, 32, 0, lane);
      oacc[dt] = wmma_bf16(ap, bv, oacc[dt]);
    }
    __syncthreads();   // all waves done with `buf` before it is refilled
  }

  // O epilogue: divide by l, write bf16 into [B][N][C]
  const int bidx = bh >> 4, hh = bh & 15;
#pragma unroll
  for (int dt = 0; dt < 4; ++dt)
#pragma unroll
    for (int i = 0; i < 8; ++i) {
      const int r = i + 8 * h;
      const int n = q0 + wave * 16 + r;
      const int c = hh * 64 + dt * 16 + (lane & 15);
      O[((size_t)bidx * NCTX + n) * CDIM + c] = (bf16)(oacc[dt][i] / l_i[i]);
    }
}

// ---------------- Kernel 4: output projection -> fp32 d_out ----------------
__global__ __launch_bounds__(256) void k_proj(const bf16* __restrict__ A, const float* __restrict__ W,
                                              const float* __restrict__ bias, float* __restrict__ out)
{
  __shared__ __align__(16) bf16 As[128][32];
  __shared__ __align__(16) bf16 Bs[128][32];
  const int tid = threadIdx.x, lane = tid & 31, wave = tid >> 5;
  const int wr = wave >> 2, wc = wave & 3;
  const int m0 = blockIdx.x * 128, n0 = blockIdx.y * 128;
  f32x8 acc[4][2] = {};
  const int ar = tid >> 2, ac = (tid & 3) * 8;
  const int bkr = tid >> 5, bc = (tid & 31) * 4;

#pragma unroll 2
  for (int k0 = 0; k0 < CDIM; k0 += 32) {
    // A tile is already bf16: pure async global->LDS copy, overlapped with W convert.
#pragma unroll
    for (int p = 0; p < 2; ++p)
      async_b128(&As[ar + 64 * p][ac], A,
                 (uint32_t)(((size_t)(m0 + ar + 64 * p) * CDIM + k0 + ac) * 2));
    if (k0 + 32 < CDIM)
      __builtin_prefetch(&W[(size_t)(k0 + 32 + bkr) * CDIM + n0 + bc], 0, 1);
#pragma unroll
    for (int p = 0; p < 4; ++p) {
      const int kk = bkr + 8 * p;
      const float4 f = *(const float4*)&W[(size_t)(k0 + kk) * CDIM + n0 + bc];
      Bs[bc + 0][kk] = (bf16)f.x; Bs[bc + 1][kk] = (bf16)f.y;
      Bs[bc + 2][kk] = (bf16)f.z; Bs[bc + 3][kk] = (bf16)f.w;
    }
    asm volatile("s_wait_asynccnt 0x0" ::: "memory");
    __syncthreads();
    const bf16x16 b0 = frag_b(&Bs[0][0], wc * 32, 32, 0, lane);
    const bf16x16 b1 = frag_b(&Bs[0][0], wc * 32 + 16, 32, 0, lane);
#pragma unroll
    for (int mt = 0; mt < 4; ++mt) {
      const bf16x16 a = frag_a(&As[0][0], wr * 64 + mt * 16, 32, 0, lane);
      acc[mt][0] = wmma_bf16(a, b0, acc[mt][0]);
      acc[mt][1] = wmma_bf16(a, b1, acc[mt][1]);
    }
    __syncthreads();
  }
  const int h = lane >> 4;
#pragma unroll
  for (int mt = 0; mt < 4; ++mt)
#pragma unroll
    for (int nt = 0; nt < 2; ++nt)
#pragma unroll
      for (int i = 0; i < 8; ++i) {
        const int row = m0 + wr * 64 + mt * 16 + i + 8 * h;
        const int col = n0 + wc * 32 + nt * 16 + (lane & 15);
        out[(size_t)row * CDIM + col] = acc[mt][nt][i] + bias[col];
      }
}

extern "C" void kernel_launch(void* const* d_in, const int* in_sizes, int n_in,
                              void* d_out, int out_size, void* d_ws, size_t ws_size,
                              hipStream_t stream) {
  const float* x     = (const float*)d_in[0];
  const float* cosf  = (const float*)d_in[1];
  const float* sinf  = (const float*)d_in[2];
  const float* Wqkv  = (const float*)d_in[3];
  const float* bqkv  = (const float*)d_in[4];
  const float* Wproj = (const float*)d_in[5];
  const float* bproj = (const float*)d_in[6];
  float* out = (float*)d_out;

  const size_t HSZ = (size_t)4 * HHEADS * NCTX * HD;   // 8388608 elems per tensor
  bf16* Qb = (bf16*)d_ws;
  bf16* Kb = Qb + HSZ;
  bf16* Vb = Kb + HSZ;                                  // [B][H][HD][N] (transposed per head)
  bf16* Ob = Vb + HSZ;                                  // [B][N][C] bf16

  k_qkv_gemm<<<dim3(64, 24), 256, 0, stream>>>(x, Wqkv, bqkv, Qb, Kb, Vb);
  k_rope<<<dim3(16384, 2), 256, 0, stream>>>(Qb, Kb, cosf, sinf);
  k_attn<<<dim3(16, 64), 256, 0, stream>>>(Qb, Kb, Vb, Ob);
  k_proj<<<dim3(64, 8), 256, 0, stream>>>(Ob, Wproj, bproj, out);
}
